// RobustFace_57638461112686
// MI455X (gfx1250) — matrix operand order repaired and checked by reading, not compiled, not run on hardware
//
#include <hip/hip_runtime.h>
#include <hip/hip_bf16.h>
#include <math.h>

typedef __attribute__((ext_vector_type(16))) _Float16 v16h;
typedef __attribute__((ext_vector_type(8)))  _Float16 v8h;
typedef __attribute__((ext_vector_type(8)))  float    v8f;

#define NUM_CLASS 100000
#define EMB       512
#define BATCH     1024
#define S_SCALE   64.0f
#define T_PLUS1   1.2f
#define CLEAR_RIO 0.9f

// trig constants (M=0.5, M1=0.2)
#define COS_M_C   0.8775825618903728f
#define SIN_M_C   0.479425538604203f
#define THETA_C   (-0.8775825618903728f)                 // cos(pi - M)
#define SINMM_C   (0.479425538604203f * 0.5f)            // sin(pi-M)*M
#define COS_M1_C  0.9800665778412416f
#define SIN_M1_C  0.19866933079506122f
#define SINMM1_C  (-0.19866933079506122f * 0.2f)         // sin(-M1)*M1

// ---------------- workspace layout (bytes) ----------------
// 0      : 2 x u32 counters (hard, noise)
// 64     : float noise_scale
// 256    : ftl  [1024 f32]
// 4352   : ftl1 [1024 f32]
// 8448   : ne16 [1024*512 f16]   (1 MB)
// 1057024: nw16 [100000*512 f16] (102.4 MB)
#define WS_CNT   0
#define WS_NS    64
#define WS_FTL   256
#define WS_FTL1  (256 + 4096)
#define WS_NE16  8448
#define WS_NW16  (8448 + (size_t)BATCH * EMB * 2)

// ---------------- kernel 1/2: row-normalize fp32 -> f16 ----------------
// one wave per row of 512; 8 rows per 256-thread block
__global__ __launch_bounds__(256)
void rf_normalize_f16(const float* __restrict__ src, _Float16* __restrict__ dst,
                      int nrows, unsigned int* __restrict__ cnt) {
  if (cnt && blockIdx.x == 0 && threadIdx.x < 2) cnt[threadIdx.x] = 0u;
  const int row  = blockIdx.x * 8 + (threadIdx.x >> 5);
  const int lane = threadIdx.x & 31;
  if (row >= nrows) return;

  const float4* p4 = (const float4*)(src + (size_t)row * EMB) + lane * 4;
  float4 a = p4[0], b = p4[1], c = p4[2], d = p4[3];
  float ss = a.x*a.x + a.y*a.y + a.z*a.z + a.w*a.w
           + b.x*b.x + b.y*b.y + b.z*b.z + b.w*b.w
           + c.x*c.x + c.y*c.y + c.z*c.z + c.w*c.w
           + d.x*d.x + d.y*d.y + d.z*d.z + d.w*d.w;
#pragma unroll
  for (int off = 16; off > 0; off >>= 1) ss += __shfl_xor(ss, off, 32);
  const float rn = rsqrtf(ss);

  v8h h0, h1;
  h0[0]=(_Float16)(a.x*rn); h0[1]=(_Float16)(a.y*rn); h0[2]=(_Float16)(a.z*rn); h0[3]=(_Float16)(a.w*rn);
  h0[4]=(_Float16)(b.x*rn); h0[5]=(_Float16)(b.y*rn); h0[6]=(_Float16)(b.z*rn); h0[7]=(_Float16)(b.w*rn);
  h1[0]=(_Float16)(c.x*rn); h1[1]=(_Float16)(c.y*rn); h1[2]=(_Float16)(c.z*rn); h1[3]=(_Float16)(c.w*rn);
  h1[4]=(_Float16)(d.x*rn); h1[5]=(_Float16)(d.y*rn); h1[6]=(_Float16)(d.z*rn); h1[7]=(_Float16)(d.w*rn);
  v8h* q = (v8h*)(dst + (size_t)row * EMB);
  q[lane * 2]     = h0;
  q[lane * 2 + 1] = h1;
}

// ---------------- kernel 3: per-row target logit -> ftl/ftl1 ----------------
// one wave per row; fp32 dot of embedding row with its label's weight row
__global__ __launch_bounds__(256)
void rf_target(const float* __restrict__ emb, const float* __restrict__ wgt,
               const int* __restrict__ labels,
               float* __restrict__ ftl, float* __restrict__ ftl1) {
  const int row  = blockIdx.x * 8 + (threadIdx.x >> 5);
  const int lane = threadIdx.x & 31;
  if (row >= BATCH) return;
  const int lab = labels[row];
  const float4* pe = (const float4*)(emb + (size_t)row * EMB) + lane * 4;
  const float4* pw = (const float4*)(wgt + (size_t)lab * EMB) + lane * 4;
  float dp = 0.f, e2 = 0.f, w2 = 0.f;
#pragma unroll
  for (int j = 0; j < 4; ++j) {
    float4 e = pe[j], w = pw[j];
    dp += e.x*w.x + e.y*w.y + e.z*w.z + e.w*w.w;
    e2 += e.x*e.x + e.y*e.y + e.z*e.z + e.w*e.w;
    w2 += w.x*w.x + w.y*w.y + w.z*w.z + w.w*w.w;
  }
#pragma unroll
  for (int off = 16; off > 0; off >>= 1) {
    dp += __shfl_xor(dp, off, 32);
    e2 += __shfl_xor(e2, off, 32);
    w2 += __shfl_xor(w2, off, 32);
  }
  if (lane == 0) {
    float tl = dp * rsqrtf(e2) * rsqrtf(w2);
    tl = fminf(1.0f, fmaxf(-1.0f, tl));
    const float st   = sqrtf(fmaxf(0.0f, 1.0f - tl * tl));
    const float ctm  = tl * COS_M_C  - st * SIN_M_C;
    const float ctm1 = tl * COS_M1_C - st * SIN_M1_C;
    ftl[row]  = (tl > THETA_C)   ? ctm  : (tl - SINMM_C);
    ftl1[row] = (tl <= COS_M1_C) ? ctm1 : (tl + SINMM1_C);
  }
}

// ---------------- kernel 4: WMMA GEMM + clip + mask counting ----------------
// block = 128 threads = 4 waves. Block tile: M=64 (shared), N=128 (32 per wave).
// Each wave: 4 M-tiles x 2 N-tiles of 16x16, K = 512 in 16 steps of 32.
__global__ __launch_bounds__(128)
void rf_gemm_wmma(const _Float16* __restrict__ A,   // ne16 [1024][512]
                  const _Float16* __restrict__ B,   // nw16 [100000][512]
                  float* __restrict__ out,          // logits [1024][100000]
                  const float* __restrict__ ftl,
                  const float* __restrict__ ftl1,
                  unsigned int* __restrict__ cnt) {
  const int lane  = threadIdx.x & 31;
  const int wave  = threadIdx.x >> 5;
  const int hl    = lane >> 4;     // half-wave select
  const int l15   = lane & 15;
  const int mBase = blockIdx.y * 64;
  const int n0    = blockIdx.x * 128 + wave * 32;
  if (n0 >= NUM_CLASS) return;     // whole-wave guard (last N block)

  v8f acc[4][2];
#pragma unroll
  for (int i = 0; i < 4; ++i)
#pragma unroll
    for (int j = 0; j < 2; ++j)
#pragma unroll
      for (int r = 0; r < 8; ++r) acc[i][j][r] = 0.0f;

  // A lane base: row = mBase + l15, contiguous 16 K at offset hl*16 within each 32-step
  const _Float16* aBase  = A + (size_t)(mBase + l15) * EMB + hl * 16;
  // B lane base: col = n0 + l15, contiguous 16 K at offset hl*16
  const _Float16* b0Base = B + (size_t)(n0 + l15) * EMB + hl * 16;
  const _Float16* b1Base = b0Base + (size_t)16 * EMB;

  for (int k0 = 0; k0 < EMB; k0 += 32) {
    const v16h b0 = *(const v16h*)(b0Base + k0);
    const v16h b1 = *(const v16h*)(b1Base + k0);
#pragma unroll
    for (int mt = 0; mt < 4; ++mt) {
      union { v16h v; v8h h[2]; } afrag;
      const _Float16* pa = aBase + (size_t)mt * 16 * EMB + k0;
      afrag.h[0] = *(const v8h*)(pa);
      afrag.h[1] = *(const v8h*)(pa + 8);
      acc[mt][0] = __builtin_amdgcn_wmma_f32_16x16x32_f16(
          false, afrag.v, false, b0, (short)0, acc[mt][0], false, false);
      acc[mt][1] = __builtin_amdgcn_wmma_f32_16x16x32_f16(
          false, afrag.v, false, b1, (short)0, acc[mt][1], false, false);
    }
  }

  // epilogue: clip, store, count masks (wave-uniform ballot counts)
  unsigned int cHard = 0, cNoise = 0;
#pragma unroll
  for (int mt = 0; mt < 4; ++mt) {
#pragma unroll
    for (int r = 0; r < 8; ++r) {
      const int row = mBase + mt * 16 + r + 8 * hl;   // C/D layout: VGPR r, halves
      const float ft  = ftl[row];
      const float ft1 = ftl1[row];
      const size_t rb = (size_t)row * NUM_CLASS + n0 + l15;
#pragma unroll
      for (int nt = 0; nt < 2; ++nt) {
        float v = acc[mt][nt][r];
        v = fminf(1.0f, fmaxf(-1.0f, v));
        cHard  += (unsigned int)__popcll(__ballot(v > ft));
        cNoise += (unsigned int)__popcll(__ballot(v > ft1));
        out[rb + nt * 16] = v;
      }
    }
  }
  if (lane == 0) {
    atomicAdd(&cnt[0], cHard);
    atomicAdd(&cnt[1], cNoise);
  }
}

// ---------------- kernel 5: global stats -> noise_scale + scalar outputs ----
__global__ void rf_stats(const unsigned int* __restrict__ cnt,
                         float* __restrict__ nsOut, float* __restrict__ tail) {
  const unsigned int hard  = cnt[0];
  const unsigned int noise = cnt[1];
  const unsigned int numAllU = (unsigned int)BATCH * (unsigned int)NUM_CLASS; // 102400000
  const unsigned int easy  = numAllU - hard;
  const float phi = ((float)easy / (float)numAllU) * 0.01f;
  const float t   = 1.0f - phi / CLEAR_RIO;
  *nsOut  = t * t;                       // GAMMA = 2
  tail[0] = (float)easy;
  tail[1] = (float)noise;
  tail[2] = (float)(hard - noise);
  tail[3] = phi;
}

// ---------------- kernel 6: in-place finalize over logits ----------------
__global__ __launch_bounds__(256)
void rf_finalize(float* __restrict__ out, const float* __restrict__ ftl,
                 const float* __restrict__ ftl1, const int* __restrict__ labels,
                 const float* __restrict__ nsPtr) {
  const int row = blockIdx.y;
  const int c0  = (blockIdx.x * 256 + threadIdx.x) * 4;
  if (c0 >= NUM_CLASS) return;
  const float ns  = *nsPtr;
  const float ft  = ftl[row];
  const float ft1 = ftl1[row];
  const int   lab = labels[row];
  float4* p = (float4*)(out + (size_t)row * NUM_CLASS + c0);
  float4 x = *p;
  float vv[4] = {x.x, x.y, x.z, x.w};
#pragma unroll
  for (int j = 0; j < 4; ++j) {
    const float v = vv[j];
    float o = v;
    if (v > ft)  o = v * T_PLUS1;                 // hard
    if (v > ft1) o = fmaxf(v * ns, 1e-30f);       // noise overrides hard
    if (c0 + j == lab) o = ft;                    // label column
    vv[j] = o * S_SCALE;
  }
  *p = make_float4(vv[0], vv[1], vv[2], vv[3]);
}

// ---------------- launcher ----------------
extern "C" void kernel_launch(void* const* d_in, const int* in_sizes, int n_in,
                              void* d_out, int out_size, void* d_ws, size_t ws_size,
                              hipStream_t stream) {
  const float* emb    = (const float*)d_in[0];  // [1024,512] f32
  const float* wgt    = (const float*)d_in[1];  // [100000,512] f32
  const int*   labels = (const int*)d_in[2];    // [1024] i32
  float* out = (float*)d_out;                    // 1024*100000 + 4 scalars

  char* ws = (char*)d_ws;
  unsigned int* cnt   = (unsigned int*)(ws + WS_CNT);
  float*        nsPtr = (float*)(ws + WS_NS);
  float*        ftl   = (float*)(ws + WS_FTL);
  float*        ftl1  = (float*)(ws + WS_FTL1);
  _Float16*     ne16  = (_Float16*)(ws + WS_NE16);
  _Float16*     nw16  = (_Float16*)(ws + WS_NW16);

  // 1) normalize embeddings -> f16 (also zeroes counters)
  rf_normalize_f16<<<BATCH / 8, 256, 0, stream>>>(emb, ne16, BATCH, cnt);
  // 2) normalize weights -> f16
  rf_normalize_f16<<<NUM_CLASS / 8, 256, 0, stream>>>(wgt, nw16, NUM_CLASS, nullptr);
  // 3) per-row target logit -> ftl/ftl1
  rf_target<<<BATCH / 8, 256, 0, stream>>>(emb, wgt, labels, ftl, ftl1);
  // 4) WMMA GEMM + clip + mask counting
  {
    const int nWaveCols = NUM_CLASS / 32;            // 3125
    dim3 grid((nWaveCols + 3) / 4, BATCH / 64);      // (782, 16)
    rf_gemm_wmma<<<grid, 128, 0, stream>>>(ne16, nw16, out, ftl, ftl1, cnt);
  }
  // 5) stats + scalar outputs
  rf_stats<<<1, 1, 0, stream>>>(cnt, nsPtr, out + (size_t)BATCH * NUM_CLASS);
  // 6) finalize logits in place
  {
    dim3 grid((NUM_CLASS / 4 + 255) / 256, BATCH);   // (98, 1024)
    rf_finalize<<<grid, 256, 0, stream>>>(out, ftl, ftl1, labels, nsPtr);
  }
}
